// Diffusion_Model_44702019616801
// MI455X (gfx1250) — compile-verified, matrix operand order checked by reading, not run on hardware
//
#include <hip/hip_runtime.h>
#include <hip/hip_bf16.h>

typedef __attribute__((ext_vector_type(16))) _Float16 v16h;
typedef __attribute__((ext_vector_type(8)))  float    v8f;

#define Tt 64
#define Nn 128
#define Hh 64
#define XPITCH 129      // x tile row pitch (floats), odd => bank-conflict free
#define H1PITCH 66      // h1 tile row pitch (halves), odd bank stride

__global__ __launch_bounds__(256)
void Diffusion_Model_44702019616801_kernel(
    const float* __restrict__ x,
    const float* __restrict__ xmean_p, const float* __restrict__ xstd_p,
    const float* __restrict__ W11, const float* __restrict__ b11,
    const float* __restrict__ g11, const float* __restrict__ be11,
    const float* __restrict__ W12, const float* __restrict__ b12,
    const float* __restrict__ W21, const float* __restrict__ b21,
    const float* __restrict__ g21, const float* __restrict__ be21,
    const float* __restrict__ W22, const float* __restrict__ b22,
    const float* __restrict__ W3,  const float* __restrict__ b3,
    const float* __restrict__ alpha_p,
    float* __restrict__ out)
{
    __shared__ float    lds_x[Tt * XPITCH];       // raw x[b] tile: [t][129] f32 (33 KB)
    __shared__ _Float16 lds_h1[Nn * H1PITCH];     // branch-1 pre-LN activations (16.5 KB)
    __shared__ float    s_h2[Hh];
    __shared__ float    s_red[Nn];
    __shared__ float    s_xd;

    const int b    = blockIdx.x;
    const int tid  = threadIdx.x;
    const int lane = tid & 31;
    const int wave = tid >> 5;

    // ---------- phase 1: async-stage x[b] (64 x 129 f32) global -> LDS ----------
    // CDNA5 direct global->LDS path (no VGPR round-trip), tracked by ASYNCcnt.
    {
        const float4* xsrc = (const float4*)(x + (size_t)b * (Tt * XPITCH));
        float4*       xdst = (float4*)lds_x;
        for (int i = tid; i < (Tt * XPITCH) / 4; i += 256) {
            const float4* gp = xsrc + i;
            unsigned ldsoff = (unsigned)(uintptr_t)(xdst + i);  // flat LDS addr[31:0] = offset
            asm volatile("global_load_async_to_lds_b128 %0, %1, off"
                         :: "v"(ldsoff), "v"(gp) : "memory");
        }
    }

    const float xmean = xmean_p[0];
    const float xinv  = 1.0f / xstd_p[0];

    asm volatile("s_wait_asynccnt 0" ::: "memory");
    __syncthreads();

    // ---------- phase 2: WMMA GEMM  h1 = u @ W11  ((128x64) @ (64x64)) ----------
    {
        const int mtile = wave;                 // rows n = mtile*16 .. mtile*16+15
        const int mrow  = lane & 15;
        const int ksel  = (lane < 16) ? 0 : 8;  // 16-bit A 16x32 VGPR layout
        const int nrow  = mtile * 16 + mrow;

        v16h a0, a1;                            // K = 0..31 and K = 32..63
        #pragma unroll
        for (int e = 0; e < 16; ++e) {
            const int kk = ksel + e + ((e >= 8) ? 8 : 0);
            a0[e] = (_Float16)((lds_x[kk * XPITCH + nrow] - xmean) * xinv);
            a1[e] = (_Float16)((lds_x[(kk + 32) * XPITCH + nrow] - xmean) * xinv);
        }

        #pragma unroll
        for (int ntile = 0; ntile < 4; ++ntile) {
            const int hcol = ntile * 16 + (lane & 15);
            v16h b0, b1;                        // B 32x16, mirrored lane layout
            #pragma unroll
            for (int e = 0; e < 16; ++e) {
                const int kk = ksel + e + ((e >= 8) ? 8 : 0);
                b0[e] = (_Float16)W11[kk * Hh + hcol];
                b1[e] = (_Float16)W11[(kk + 32) * Hh + hcol];
            }
            v8f c = {};
            c = __builtin_amdgcn_wmma_f32_16x16x32_f16(false, a0, false, b0,
                                                       (short)0, c, false, false);
            c = __builtin_amdgcn_wmma_f32_16x16x32_f16(false, a1, false, b1,
                                                       (short)0, c, false, false);
            const float bias = b11[hcol];
            #pragma unroll
            for (int r = 0; r < 8; ++r) {       // C layout: VGPR r -> M = r (+8 for hi lanes)
                const int m = mtile * 16 + r + ((lane >> 4) << 3);
                lds_h1[m * H1PITCH + hcol] = (_Float16)(c[r] + bias);
            }
        }
    }
    __syncthreads();

    // ---------- phase 3: down-branch (identical for all n) on threads 0..63 ----------
    if (tid < Hh) {
        float acc = b21[tid];
        for (int k = 0; k < Tt; ++k) {
            const float dk = (lds_x[k * XPITCH + Nn] - xmean) * xinv;
            acc += dk * W21[k * Hh + tid];
        }
        s_h2[tid] = acc;
    }
    __syncthreads();
    if (tid < Hh) {
        float mu = 0.f;
        for (int h = 0; h < Hh; ++h) mu += s_h2[h];
        mu *= (1.0f / Hh);
        float var = 0.f;
        for (int h = 0; h < Hh; ++h) { const float d = s_h2[h] - mu; var += d * d; }
        var *= (1.0f / Hh);
        float hv = (s_h2[tid] - mu) * rsqrtf(var + 1e-5f) * g21[tid] + be21[tid];
        hv = fmaxf(hv, 0.f);
        s_red[tid] = hv * W22[tid];
    }
    __syncthreads();
    if (tid == 0) {
        float acc = b22[0];
        for (int h = 0; h < Hh; ++h) acc += s_red[h];
        s_xd = 1.0f / (1.0f + __expf(-acc));
    }
    __syncthreads();

    // ---------- phase 4: per-row LN + head + horizon-weighted sum (threads 0..127) ----------
    float part = 0.f;
    if (tid < Nn) {
        const int n = tid;
        float mu = 0.f;
        for (int h = 0; h < Hh; ++h) mu += (float)lds_h1[n * H1PITCH + h];
        mu *= (1.0f / Hh);
        float var = 0.f;
        for (int h = 0; h < Hh; ++h) {
            const float d = (float)lds_h1[n * H1PITCH + h] - mu; var += d * d;
        }
        var *= (1.0f / Hh);
        const float rstd = rsqrtf(var + 1e-5f);
        float acc = b12[0];
        for (int h = 0; h < Hh; ++h) {
            float hv = ((float)lds_h1[n * H1PITCH + h] - mu) * rstd * g11[h] + be11[h];
            hv = fmaxf(hv, 0.f);
            acc += hv * W12[h];
        }
        const float xu = 1.0f / (1.0f + __expf(-acc));
        const float v  = xu * W3[0] + s_xd * W3[1] + b3[0];
        const float Tv = 50.0f / (v + 1e-5f);
        const float F  = 1.0f / (1.0f + alpha_p[0] * Tv);
        float Tidx = rintf(Tv * (1.0f / 10.0f));           // jnp.round = RNE
        if (Tidx < 0.f)        Tidx = 0.f;
        if (Tidx > (float)Tt)  Tidx = (float)Tt - 1.0f;
        const int Lni = (int)fmaxf((float)Tt - Tidx, 1.0f); // integer-valued by construction
        const float omF = 1.0f - F;
        float w = F;                                        // j = Lni-1 has exponent 0
        for (int j = Lni - 1; j >= 0; --j) {
            part += w * lds_x[j * XPITCH + n];
            w *= omF;
        }
    }
    __syncthreads();            // s_red reuse: phase-3 writes are done
    if (tid < Nn) s_red[tid] = part;
    __syncthreads();
    if (tid == 0) {
        float acc = 0.f;
        for (int n = 0; n < Nn; ++n) acc += s_red[n];
        out[b] = acc;
    }
}

extern "C" void kernel_launch(void* const* d_in, const int* in_sizes, int n_in,
                              void* d_out, int out_size, void* d_ws, size_t ws_size,
                              hipStream_t stream) {
    (void)in_sizes; (void)n_in; (void)d_ws; (void)ws_size; (void)out_size;
    const float* xx    = (const float*)d_in[0];
    const float* xmean = (const float*)d_in[1];
    const float* xstd  = (const float*)d_in[2];
    const float* W11   = (const float*)d_in[3];
    const float* b11   = (const float*)d_in[4];
    const float* g11   = (const float*)d_in[5];
    const float* be11  = (const float*)d_in[6];
    const float* W12   = (const float*)d_in[7];
    const float* b12   = (const float*)d_in[8];
    const float* W21   = (const float*)d_in[9];
    const float* b21   = (const float*)d_in[10];
    const float* g21   = (const float*)d_in[11];
    const float* be21  = (const float*)d_in[12];
    const float* W22   = (const float*)d_in[13];
    const float* b22   = (const float*)d_in[14];
    const float* W3    = (const float*)d_in[15];
    const float* b3    = (const float*)d_in[16];
    const float* alpha = (const float*)d_in[17];
    float* out = (float*)d_out;

    Diffusion_Model_44702019616801_kernel<<<4096, 256, 0, stream>>>(
        xx, xmean, xstd, W11, b11, g11, be11, W12, b12,
        W21, b21, g21, be21, W22, b22, W3, b3, alpha, out);
}
